// Llama4TextExperts_6863357739469
// MI455X (gfx1250) — matrix operand order repaired and checked by reading, not compiled, not run on hardware
//
#include <hip/hip_runtime.h>
#include <hip/hip_bf16.h>

typedef __bf16 bf16;
typedef __attribute__((ext_vector_type(16))) __bf16 v16bf;
typedef __attribute__((ext_vector_type(8)))  __bf16 v8bf;
typedef __attribute__((ext_vector_type(8)))  float  v8f;
typedef __attribute__((ext_vector_type(4)))  float  f4;

#define HID     2048      // hidden size H
#define DEXP    2048      // expert dim D
#define TM      64        // tokens per workgroup tile
#define A_PITCH 80        // bytes: 32 bf16 + pad (16 rows -> 16 distinct banks)
#define B_PITCH 80        // bytes: transposed (column-major) staging, K contiguous
#define ACT_PITCH 4176    // bytes: 2048 bf16 + pad (4176/4 % 64 == 20 -> bank spread)
#define A_BYTES (TM * A_PITCH)              // 5120
#define B_BYTES (256 * B_PITCH)             // 20480
#define ACT_OFF 0
#define A_OFF   (TM * ACT_PITCH)            // 267264
#define B_OFF   (A_OFF + 2 * A_BYTES)       // 277504
#define LDS_BYTES (B_OFF + 2 * B_BYTES)     // 318464  (< 320 KB WGP LDS)

struct bfpair { v8bf lo, hi; };

static __device__ __forceinline__ v16bf make16(v8bf lo, v8bf hi) {
    bfpair p{lo, hi};
    return __builtin_bit_cast(v16bf, p);
}

static __device__ __forceinline__ unsigned short bfbits(float f) {
    return __builtin_bit_cast(unsigned short, (bf16)f);
}

static __device__ __forceinline__ unsigned pack2(float a, float b) {
    return (unsigned)bfbits(a) | ((unsigned)bfbits(b) << 16);
}

static __device__ __forceinline__ unsigned long long pack4(float a, float b,
                                                           float c, float d) {
    return (unsigned long long)pack2(a, b) | ((unsigned long long)pack2(c, d) << 32);
}

__global__ __launch_bounds__(256, 1)
void moe_swiglu_fused(const float* __restrict__ x,
                      const float* __restrict__ gup,
                      const float* __restrict__ dwn,
                      float* __restrict__ out)
{
    extern __shared__ unsigned char lds[];

    const int tid  = threadIdx.x;
    const int lane = tid & 31;
    const int wave = tid >> 5;       // 0..7
    const int wm   = wave >> 2;      // 0..1 : 32-row half (2 M-tiles per wave)
    const int wn   = wave & 3;       // 0..3 : 32-col slice (2 N-tiles per wave)
    const int l15  = lane & 15;
    const int aHalf = (lane & 16) ? 8  : 0;   // A-frag K-half base (elements)
    const int bHalf = (lane & 16) ? 8  : 0;   // B-frag K-half base (elements)
    const int rOff  = (lane & 16) ? 8  : 0;   // C/D row offset for hi lanes

    const int e    = blockIdx.x >> 6;         // expert-major for L2 weight reuse
    const int tile = blockIdx.x & 63;
    const size_t rowBase = (size_t)e * 4096 + (size_t)tile * TM;
    const float* X  = x   + rowBase * HID;
    const float* W1 = gup + (size_t)e * HID * (2 * DEXP);
    const float* W2 = dwn + (size_t)e * DEXP * HID;
    float* O = out + rowBase * HID;

    const int ar  = tid >> 2;        // A-staging row
    const int acg = tid & 3;         // A-staging col group (8 floats)

    // ================= Phase 1: gate_up GEMM + SwiGLU -> act in LDS =================
    for (int nc = 0; nc < 16; ++nc) {
        const int gcol0 = nc * 128;
        v8f accG[2][2], accU[2][2];
        #pragma unroll
        for (int i = 0; i < 2; ++i)
            #pragma unroll
            for (int j = 0; j < 2; ++j) { accG[i][j] = (v8f)0.0f; accU[i][j] = (v8f)0.0f; }

        f4 ra0, ra1, rb[2][4];

        // ---- prologue: stage K-step 0 into buffer 0 ----
        {
            const float* s = X + (size_t)ar * HID + acg * 8;
            ra0 = *(const f4*)s; ra1 = *(const f4*)(s + 4);
            #pragma unroll
            for (int rep = 0; rep < 2; ++rep) {
                const int g = rep * 256 + tid;
                const int k = (g >> 6) * 4, c = (g & 63) * 4;
                const int gc = (c < 128) ? (gcol0 + c) : (2048 + gcol0 + (c - 128));
                const float* ws = W1 + (size_t)k * (2 * DEXP) + gc;
                #pragma unroll
                for (int j = 0; j < 4; ++j) rb[rep][j] = *(const f4*)(ws + (size_t)j * 2 * DEXP);
            }
            v8bf pa;
            #pragma unroll
            for (int u = 0; u < 4; ++u) { pa[u] = (bf16)ra0[u]; pa[4 + u] = (bf16)ra1[u]; }
            *(v8bf*)(lds + A_OFF + ar * A_PITCH + acg * 16) = pa;
            #pragma unroll
            for (int rep = 0; rep < 2; ++rep) {
                const int g = rep * 256 + tid;
                const int k = (g >> 6) * 4, c = (g & 63) * 4;
                #pragma unroll
                for (int u = 0; u < 4; ++u)
                    *(unsigned long long*)(lds + B_OFF + (c + u) * B_PITCH + k * 2) =
                        pack4(rb[rep][0][u], rb[rep][1][u], rb[rep][2][u], rb[rep][3][u]);
            }
        }

        for (int kk = 0; kk < 64; ++kk) {
            const int p = kk & 1;
            __syncthreads();                       // buf p ready; buf p^1 drained
            const bool more = (kk < 63);
            if (more) {                            // issue next K-step's global loads
                const int k0n = (kk + 1) * 32;
                const float* s = X + (size_t)ar * HID + k0n + acg * 8;
                ra0 = *(const f4*)s; ra1 = *(const f4*)(s + 4);
                #pragma unroll
                for (int rep = 0; rep < 2; ++rep) {
                    const int g = rep * 256 + tid;
                    const int k = (g >> 6) * 4, c = (g & 63) * 4;
                    const int gc = (c < 128) ? (gcol0 + c) : (2048 + gcol0 + (c - 128));
                    const float* ws = W1 + (size_t)(k0n + k) * (2 * DEXP) + gc;
                    #pragma unroll
                    for (int j = 0; j < 4; ++j) rb[rep][j] = *(const f4*)(ws + (size_t)j * 2 * DEXP);
                }
            }
            // ---- WMMA compute from buffer p (overlaps the loads above) ----
            const unsigned char* abuf = lds + A_OFF + p * A_BYTES;
            const unsigned char* bbuf = lds + B_OFF + p * B_BYTES;
            v16bf af[2];
            #pragma unroll
            for (int mt = 0; mt < 2; ++mt) {
                const unsigned char* ab = abuf + (wm * 32 + mt * 16 + l15) * A_PITCH;
                af[mt] = make16(*(const v8bf*)(ab + aHalf * 2),
                                *(const v8bf*)(ab + (aHalf + 16) * 2));
            }
            #pragma unroll
            for (int nt = 0; nt < 2; ++nt) {
                const int cG = wn * 32 + nt * 16 + l15;
                const unsigned char* bbG = bbuf + cG * B_PITCH + bHalf * 2;
                v16bf bG = make16(*(const v8bf*)bbG, *(const v8bf*)(bbG + 32));
                const unsigned char* bbU = bbG + 128 * B_PITCH;
                v16bf bU = make16(*(const v8bf*)bbU, *(const v8bf*)(bbU + 32));
                #pragma unroll
                for (int mt = 0; mt < 2; ++mt) {
                    accG[mt][nt] = __builtin_amdgcn_wmma_f32_16x16x32_bf16(
                        false, af[mt], false, bG, (short)0, accG[mt][nt], false, false);
                    accU[mt][nt] = __builtin_amdgcn_wmma_f32_16x16x32_bf16(
                        false, af[mt], false, bU, (short)0, accU[mt][nt], false, false);
                }
            }
            if (more) {                            // commit next step into buffer p^1
                unsigned char* abufn = lds + A_OFF + (p ^ 1) * A_BYTES;
                unsigned char* bbufn = lds + B_OFF + (p ^ 1) * B_BYTES;
                v8bf pa;
                #pragma unroll
                for (int u = 0; u < 4; ++u) { pa[u] = (bf16)ra0[u]; pa[4 + u] = (bf16)ra1[u]; }
                *(v8bf*)(abufn + ar * A_PITCH + acg * 16) = pa;
                #pragma unroll
                for (int rep = 0; rep < 2; ++rep) {
                    const int g = rep * 256 + tid;
                    const int k = (g >> 6) * 4, c = (g & 63) * 4;
                    #pragma unroll
                    for (int u = 0; u < 4; ++u)
                        *(unsigned long long*)(bbufn + (c + u) * B_PITCH + k * 2) =
                            pack4(rb[rep][0][u], rb[rep][1][u], rb[rep][2][u], rb[rep][3][u]);
                }
            }
        }
        // ---- SwiGLU epilogue: act = up * g * rcp(1+exp(-g)), bf16 -> LDS act ----
        #pragma unroll
        for (int mt = 0; mt < 2; ++mt) {
            #pragma unroll
            for (int nt = 0; nt < 2; ++nt) {
                #pragma unroll
                for (int v = 0; v < 8; ++v) {
                    const float gv = accG[mt][nt][v];
                    const float uv = accU[mt][nt][v];
                    const float sig = __builtin_amdgcn_rcpf(1.0f + __expf(-gv));
                    const float av  = uv * gv * sig;
                    const int row = wm * 32 + mt * 16 + v + rOff;
                    const int col = gcol0 + wn * 32 + nt * 16 + l15;
                    *(unsigned short*)(lds + ACT_OFF + row * ACT_PITCH + col * 2) = bfbits(av);
                }
            }
        }
    }

    // ================= Phase 2: out = act @ W2, A straight from LDS =================
    for (int hc = 0; hc < 16; ++hc) {
        const int hbase = hc * 128;
        v8f acc[2][2];
        #pragma unroll
        for (int i = 0; i < 2; ++i)
            #pragma unroll
            for (int j = 0; j < 2; ++j) acc[i][j] = (v8f)0.0f;

        f4 rb[4];
        const int bk = (tid >> 5) * 4;   // staged K-quad row
        const int bc = (tid & 31) * 4;   // staged col group

        // ---- prologue: stage K-step 0 into buffer 0 ----
        {
            #pragma unroll
            for (int j = 0; j < 4; ++j)
                rb[j] = *(const f4*)(W2 + (size_t)(bk + j) * HID + hbase + bc);
            #pragma unroll
            for (int u = 0; u < 4; ++u)
                *(unsigned long long*)(lds + B_OFF + (bc + u) * B_PITCH + bk * 2) =
                    pack4(rb[0][u], rb[1][u], rb[2][u], rb[3][u]);
        }

        for (int kk = 0; kk < 64; ++kk) {
            const int p = kk & 1;
            const int k0 = kk * 32;
            __syncthreads();                       // also orders phase-1 act stores
            const bool more = (kk < 63);
            if (more) {
                #pragma unroll
                for (int j = 0; j < 4; ++j)
                    rb[j] = *(const f4*)(W2 + (size_t)(k0 + 32 + bk + j) * HID + hbase + bc);
            }
            const unsigned char* bbuf = lds + B_OFF + p * B_BYTES;
            v16bf af[2];
            #pragma unroll
            for (int mt = 0; mt < 2; ++mt) {
                const unsigned char* ab =
                    lds + ACT_OFF + (wm * 32 + mt * 16 + l15) * ACT_PITCH + k0 * 2;
                af[mt] = make16(*(const v8bf*)(ab + aHalf * 2),
                                *(const v8bf*)(ab + (aHalf + 16) * 2));
            }
            #pragma unroll
            for (int nt = 0; nt < 2; ++nt) {
                const int cb = wn * 32 + nt * 16 + l15;
                const unsigned char* bb = bbuf + cb * B_PITCH + bHalf * 2;
                v16bf b = make16(*(const v8bf*)bb, *(const v8bf*)(bb + 32));
                #pragma unroll
                for (int mt = 0; mt < 2; ++mt)
                    acc[mt][nt] = __builtin_amdgcn_wmma_f32_16x16x32_bf16(
                        false, af[mt], false, b, (short)0, acc[mt][nt], false, false);
            }
            if (more) {
                unsigned char* bbufn = lds + B_OFF + (p ^ 1) * B_BYTES;
                #pragma unroll
                for (int u = 0; u < 4; ++u)
                    *(unsigned long long*)(bbufn + (bc + u) * B_PITCH + bk * 2) =
                        pack4(rb[0][u], rb[1][u], rb[2][u], rb[3][u]);
            }
        }
        // ---- write fp32 output (lanes cover 16 consecutive cols: coalesced) ----
        #pragma unroll
        for (int mt = 0; mt < 2; ++mt) {
            #pragma unroll
            for (int nt = 0; nt < 2; ++nt) {
                #pragma unroll
                for (int v = 0; v < 8; ++v) {
                    const int row = wm * 32 + mt * 16 + v + rOff;
                    const int col = hbase + wn * 32 + nt * 16 + l15;
                    O[(size_t)row * HID + col] = acc[mt][nt][v];
                }
            }
        }
    }
}

extern "C" void kernel_launch(void* const* d_in, const int* in_sizes, int n_in,
                              void* d_out, int out_size, void* d_ws, size_t ws_size,
                              hipStream_t stream) {
    const float* x   = (const float*)d_in[0];  // (32768, 2048)
    const float* gup = (const float*)d_in[1];  // (8, 2048, 4096)
    const float* dwn = (const float*)d_in[2];  // (8, 2048, 2048)
    float* out = (float*)d_out;                // (32768, 2048)
    (void)in_sizes; (void)n_in; (void)out_size; (void)d_ws; (void)ws_size;

    dim3 grid(512);   // 8 experts * 64 token-tiles, expert-major
    dim3 block(256);  // 8 waves (wave32)
    moe_swiglu_fused<<<grid, block, LDS_BYTES, stream>>>(x, gup, dwn, out);
}
